// PG_few_11776800325981
// MI455X (gfx1250) — matrix operand order repaired and checked by reading, not compiled, not run on hardware
//
#include <hip/hip_runtime.h>
#include <hip/hip_bf16.h>

#define N_ROWS 50000
#define DIM    384
#define KP     8
#define BATCH  512
#define NEGO   128
#define EDGES  1024
#define NEG_INF -1e9f

typedef __attribute__((ext_vector_type(16))) __bf16 v16bf;
typedef __attribute__((ext_vector_type(8)))  float  v8f;

// ---------------- workspace layout (bytes) ----------------
// [0,       12288)  Vn   : 8*384 f32 (normalized virtual)
// [12288,   12544)  Ginv : 64 f32
// [12544,   14592)  score_raw : 512 f32
// [16384,   ...  )  local bf16 : 50000*384 u16  (38.4 MB)
#define WS_VN     0
#define WS_GINV   12288
#define WS_SCORE  12544
#define WS_LOCAL  16384

__device__ inline float wred_sum(float v) {
  #pragma unroll
  for (int o = 16; o; o >>= 1) v += __shfl_xor(v, o, 32);
  return v;
}
__device__ inline float wred_max(float v) {
  #pragma unroll
  for (int o = 16; o; o >>= 1) v = fmaxf(v, __shfl_xor(v, o, 32));
  return v;
}
__device__ inline unsigned short f2bf(float f) {
  unsigned u = __builtin_bit_cast(unsigned, f);
  u += 0x7FFFu + ((u >> 16) & 1u);        // round to nearest even
  return (unsigned short)(u >> 16);
}
__device__ inline __bf16 bfbits(unsigned short s) {
  return __builtin_bit_cast(__bf16, s);
}

// ================= kernel 1: V-normalize, G, G^-1 =================
__global__ void k1_prep(const float* __restrict__ virt, float* __restrict__ ws_vn,
                        float* __restrict__ ws_ginv) {
  __shared__ float sV[KP * DIM];
  __shared__ float sG[KP * 16];     // augmented [G | I]
  const int t = threadIdx.x, lane = t & 31, w = t >> 5;

  // each of 8 waves normalizes one row of virtual
  {
    float ss = 0.f;
    float e[12];
    #pragma unroll
    for (int i = 0; i < 12; ++i) { e[i] = virt[w * DIM + lane + 32 * i]; ss += e[i] * e[i]; }
    ss = wred_sum(ss);
    float inv = rsqrtf(ss + 1e-12f);
    #pragma unroll
    for (int i = 0; i < 12; ++i) {
      float v = e[i] * inv;
      sV[w * DIM + lane + 32 * i] = v;
      ws_vn[w * DIM + lane + 32 * i] = v;
    }
  }
  __syncthreads();

  if (t < 64) {                      // G = Vn Vn^T + eps I, build augmented
    int k = t >> 3, k2 = t & 7;
    float d = 0.f;
    for (int j = 0; j < DIM; ++j) d += sV[k * DIM + j] * sV[k2 * DIM + j];
    sG[k * 16 + k2] = d + ((k == k2) ? 1e-4f : 0.f);
    sG[k * 16 + 8 + k2] = (k == k2) ? 1.f : 0.f;
  }
  __syncthreads();

  if (t == 0) {                      // Gauss-Jordan (G is SPD, no pivoting)
    for (int p = 0; p < KP; ++p) {
      float ip = 1.f / sG[p * 16 + p];
      for (int c = 0; c < 16; ++c) sG[p * 16 + c] *= ip;
      for (int r = 0; r < KP; ++r) if (r != p) {
        float f = sG[r * 16 + p];
        for (int c = 0; c < 16; ++c) sG[r * 16 + c] -= f * sG[p * 16 + c];
      }
    }
  }
  __syncthreads();
  if (t < 64) ws_ginv[t] = sG[(t >> 3) * 16 + 8 + (t & 7)];
}

// ========= kernel 2: per-row projection removal -> bf16 local =========
__global__ void k2_local(const float* __restrict__ emb, const float* __restrict__ ws_vn,
                         const float* __restrict__ ws_ginv, unsigned short* __restrict__ localbf) {
  __shared__ float sVn[KP * DIM];
  __shared__ float sGi[64];
  const int t = threadIdx.x, lane = t & 31, w = t >> 5;
  for (int i = t; i < KP * DIM; i += 256) sVn[i] = ws_vn[i];
  if (t < 64) sGi[t] = ws_ginv[t];
  __syncthreads();

  const int row = blockIdx.x * 8 + w;
  if (row >= N_ROWS) return;

  float e[12];
  float ss = 0.f;
  #pragma unroll
  for (int i = 0; i < 12; ++i) { e[i] = emb[(size_t)row * DIM + lane + 32 * i]; ss += e[i] * e[i]; }
  ss = wred_sum(ss);
  const float inv = rsqrtf(ss + 1e-12f);

  float acc[KP];
  #pragma unroll
  for (int k = 0; k < KP; ++k) acc[k] = 0.f;
  #pragma unroll
  for (int i = 0; i < 12; ++i) {
    const int j = lane + 32 * i;
    const float ev = e[i];
    #pragma unroll
    for (int k = 0; k < KP; ++k) acc[k] += ev * sVn[k * DIM + j];
  }
  float c[KP], coef[KP];
  #pragma unroll
  for (int k = 0; k < KP; ++k) c[k] = wred_sum(acc[k]) * inv;   // (emb_n . V^T)
  #pragma unroll
  for (int k2 = 0; k2 < KP; ++k2) {
    float s = 0.f;
    #pragma unroll
    for (int k = 0; k < KP; ++k) s += c[k] * sGi[k * 8 + k2];   // coef = c G^-1
    coef[k2] = s;
  }

  float o[12], ss2 = 0.f;
  #pragma unroll
  for (int i = 0; i < 12; ++i) {
    const int j = lane + 32 * i;
    float v = e[i] * inv;
    #pragma unroll
    for (int k = 0; k < KP; ++k) v -= coef[k] * sVn[k * DIM + j];
    o[i] = v; ss2 += v * v;
  }
  ss2 = wred_sum(ss2);
  const float inv2 = rsqrtf(ss2 + 1e-12f);
  #pragma unroll
  for (int i = 0; i < 12; ++i)
    localbf[(size_t)row * DIM + lane + 32 * i] = f2bf(o[i] * inv2);
}

// ====== kernel 3: per-ego-graph bf16 WMMA GEMM + softmax + diff ======
#define BSTRIDE 400                           // u16 stride (800B: 32B-aligned, bank-rotated)
#define SM_BATCH_BYTES (NEGO * BSTRIDE * 2)   // 102400; raw sim (64KB f32) overlays this
#define SM_BITMAP_OFF  SM_BATCH_BYTES
#define SM_DIFF_OFF    (SM_BITMAP_OFF + NEGO * 4 * 4)
#define SM_TOTAL       (SM_DIFF_OFF + NEGO * 4)

__global__ __launch_bounds__(256)
void k3_ego(const unsigned short* __restrict__ localbf, const float* __restrict__ gumbel,
            const int* __restrict__ idx_batch, const int* __restrict__ esrc,
            const int* __restrict__ edst, float* __restrict__ score_raw) {
  extern __shared__ __align__(64) char smem[];
  unsigned short* sB   = (unsigned short*)smem;           // batch bf16 [128][400]
  float*          sSim = (float*)smem;                    // overlays sB after GEMM
  unsigned*       sAdj = (unsigned*)(smem + SM_BITMAP_OFF);
  float*          sDiff = (float*)(smem + SM_DIFF_OFF);

  const int b = blockIdx.x;
  const int t = threadIdx.x, lane = t & 31, w = t >> 5, hi = lane >> 4;

  // gather 128 ego rows (as u32 pairs of bf16) + zero/scatter adjacency bitmap
  for (int i = t; i < NEGO * 4; i += 256) sAdj[i] = 0u;
  for (int e = t; e < NEGO * (DIM / 2); e += 256) {
    const int row = e / (DIM / 2), c2 = e % (DIM / 2);
    const int idx = idx_batch[b * NEGO + row];
    const unsigned* src = (const unsigned*)(localbf + (size_t)idx * DIM);
    ((unsigned*)(sB + row * BSTRIDE))[c2] = src[c2];
  }
  __syncthreads();
  for (int i = t; i < EDGES; i += 256) {
    const int s = esrc[b * EDGES + i], d = edst[b * EDGES + i];
    atomicOr(&sAdj[s * 4 + (d >> 5)], 1u << (d & 31));
  }
  // pull this block's gumbel tile (64KB) toward L2 while the matrix pipes run
  {
    const float* gb = gumbel + (size_t)b * NEGO * NEGO;
    for (int i = t; i < 512; i += 256) __builtin_prefetch(gb + i * 32, 0, 1);
  }

  // GEMM: wave w owns a 32-row x 64-col block: 2 row tiles x 4 col tiles.
  // Per k-step: 2 A frags + 4 B frags = 12 ds_load_b128 feeding 8 WMMAs.
  v8f acc[8] = {};                    // acc[j*2 + rt]
  const int rowBase = (w & 3) * 32;
  const int colBase = (w >> 2) * 64;
  const unsigned short* arow0 = sB + (rowBase + (lane & 15)) * BSTRIDE;
  const unsigned short* arow1 = arow0 + 16 * BSTRIDE;
  const unsigned short* brow  = sB + (colBase + (lane & 15)) * BSTRIDE + hi * 16;
  for (int kb = 0; kb < DIM; kb += 32) {
    v16bf a0, a1;
    #pragma unroll
    for (int v = 0; v < 8; ++v) {     // A 16x32: 16 contiguous K per lane (k = 16*hi + s), matches B
      const int kk = kb + 16 * hi + 2 * v;
      a0[2 * v]     = bfbits(arow0[kk]);
      a0[2 * v + 1] = bfbits(arow0[kk + 1]);
      a1[2 * v]     = bfbits(arow1[kk]);
      a1[2 * v + 1] = bfbits(arow1[kk + 1]);
    }
    #pragma unroll
    for (int j = 0; j < 4; ++j) {     // B 32x16: 16 contiguous K per lane, n = lane&15
      const v16bf bb = *reinterpret_cast<const v16bf*>(brow + j * 16 * BSTRIDE + kb);
      acc[j * 2]     = __builtin_amdgcn_wmma_f32_16x16x32_bf16(
          false, a0, false, bb, (short)0, acc[j * 2], false, false);
      acc[j * 2 + 1] = __builtin_amdgcn_wmma_f32_16x16x32_bf16(
          false, a1, false, bb, (short)0, acc[j * 2 + 1], false, false);
    }
  }
  __syncthreads();                    // done reading sB; overlay with raw sim

  #pragma unroll
  for (int j = 0; j < 4; ++j) {       // C/D layout: VGPR r -> M=tileRow+r+8*hi, N=lane&15
    const int n = colBase + j * 16 + (lane & 15);
    #pragma unroll
    for (int rt = 0; rt < 2; ++rt) {
      const int mb = rowBase + rt * 16 + 8 * hi;
      #pragma unroll
      for (int r = 0; r < 8; ++r) sSim[(mb + r) * NEGO + n] = acc[j * 2 + rt][r];
    }
  }
  __syncthreads();

  // fused gumbel-softmax + ||adj - p|| per row (wave w owns rows w, w+8, ...)
  for (int r = w; r < NEGO; r += 8) {
    const float* grow = gumbel + ((size_t)b * NEGO + r) * NEGO;
    float x[4], mx = -3.4e38f;
    #pragma unroll
    for (int i = 0; i < 4; ++i) {
      const int m = lane + 32 * i;
      const float sv = (m == r) ? NEG_INF : sSim[r * NEGO + m];
      x[i] = sv + grow[m];
      mx = fmaxf(mx, x[i]);
    }
    mx = wred_max(mx);
    float sum = 0.f;
    #pragma unroll
    for (int i = 0; i < 4; ++i) { x[i] = __expf(x[i] - mx); sum += x[i]; }
    sum = wred_sum(sum);
    const float invs = 1.f / sum;
    float d2 = 0.f;
    #pragma unroll
    for (int i = 0; i < 4; ++i) {
      const int m = lane + 32 * i;
      const float p = x[i] * invs;
      const float a = (float)((sAdj[r * 4 + (m >> 5)] >> (m & 31)) & 1u);
      const float dd = a - p;
      d2 += dd * dd;
    }
    d2 = wred_sum(d2);
    if (lane == 0) sDiff[r] = sqrtf(d2);
  }
  __syncthreads();

  if (t < 32) {
    float s = 0.f;
    for (int i = lane; i < NEGO; i += 32) s += sDiff[i];
    s = wred_sum(s);
    if (lane == 0) score_raw[b] = s * (1.f / (float)NEGO);
  }
}

// ============ kernel 4: min-max normalize + BCE loss ============
__global__ void k4_score(const float* __restrict__ score_raw, const int* __restrict__ label,
                         float* __restrict__ out) {
  __shared__ float red[BATCH];
  const int t = threadIdx.x;
  const float s = score_raw[t];

  red[t] = s; __syncthreads();
  for (int o = 256; o; o >>= 1) { if (t < o) red[t] = fminf(red[t], red[t + o]); __syncthreads(); }
  const float mn = red[0]; __syncthreads();
  red[t] = s; __syncthreads();
  for (int o = 256; o; o >>= 1) { if (t < o) red[t] = fmaxf(red[t], red[t + o]); __syncthreads(); }
  const float mx = red[0]; __syncthreads();

  const float sn = (s - mn) / (mx - mn + 1e-8f);
  out[t] = sn;
  const float y = (float)label[t];
  const float lp  = fmaxf(logf(sn), -100.f);
  const float l1p = fmaxf(log1pf(-sn), -100.f);
  red[t] = -(y * lp + (1.f - y) * l1p); __syncthreads();
  for (int o = 256; o; o >>= 1) { if (t < o) red[t] += red[t + o]; __syncthreads(); }
  if (t == 0) out[BATCH] = red[0] * (1.f / (float)BATCH);
}

extern "C" void kernel_launch(void* const* d_in, const int* in_sizes, int n_in,
                              void* d_out, int out_size, void* d_ws, size_t ws_size,
                              hipStream_t stream) {
  const float* text_emb = (const float*)d_in[0];   // [50000,384]
  const float* virt     = (const float*)d_in[1];   // [8,384]
  const float* gumbel   = (const float*)d_in[2];   // [512,128,128]
  const int*   idxb     = (const int*)d_in[3];     // [512,128]
  const int*   esrc     = (const int*)d_in[4];     // [512,1024]
  const int*   edst     = (const int*)d_in[5];     // [512,1024]
  const int*   label    = (const int*)d_in[6];     // [512]
  float* out = (float*)d_out;                      // [513] = score[512], loss

  char* ws = (char*)d_ws;
  float* ws_vn    = (float*)(ws + WS_VN);
  float* ws_ginv  = (float*)(ws + WS_GINV);
  float* ws_score = (float*)(ws + WS_SCORE);
  unsigned short* ws_local = (unsigned short*)(ws + WS_LOCAL);

  (void)hipFuncSetAttribute((const void*)k3_ego,
                            hipFuncAttributeMaxDynamicSharedMemorySize, SM_TOTAL);

  k1_prep <<<1, 256, 0, stream>>>(virt, ws_vn, ws_ginv);
  k2_local<<<(N_ROWS + 7) / 8, 256, 0, stream>>>(text_emb, ws_vn, ws_ginv, ws_local);
  k3_ego  <<<BATCH, 256, SM_TOTAL, stream>>>(ws_local, gumbel, idxb, esrc, edst, ws_score);
  k4_score<<<1, BATCH, 0, stream>>>(ws_score, label, out);
}
